// LSTMPredictor_44143673869018
// MI455X (gfx1250) — matrix operand order repaired and checked by reading, not compiled, not run on hardware
//
#include <hip/hip_runtime.h>

typedef __attribute__((ext_vector_type(16))) _Float16 v16h;
typedef __attribute__((ext_vector_type(4)))  _Float16 v4h;
typedef __attribute__((ext_vector_type(8)))  float    v8f;

#define HN      128
#define BN      512
#define TN      1024
#define FUTN    64
#define TILE_B  16
#define GPAD    20      // padded row length (floats) for transposed gate buffer
#define NTHREADS 512    // 16 waves (wave32): 256-VGPR budget fits 192 VGPRs of weights

__device__ __forceinline__ float fsig(float x) {
  return __builtin_amdgcn_rcpf(1.0f + __expf(-x));   // v_exp_f32 + v_rcp_f32
}
__device__ __forceinline__ float ftanh(float x) { return 2.0f * fsig(2.0f * x) - 1.0f; }

// read 16 consecutive f32 weights, convert to an f16 WMMA B-fragment slice
__device__ __forceinline__ v16h load_w_frag(const float* __restrict__ p) {
  v16h r;
#pragma unroll
  for (int i = 0; i < 16; ++i) r[i] = (_Float16)p[i];
  return r;
}

struct __align__(32) Smem {
  float    gbufT[512][GPAD];        // 40KB transposed gate tile: [gate col n][batch b]
  _Float16 hbuf1[4][32][16];        // 4KB  h1 in A-fragment lane layout [kchunk][lane][half]
  _Float16 hbuf2[4][32][16];        // 4KB  h2 in A-fragment lane layout (also projection src)
  float    bias1[512];              // b_ih1 + b_hh1
  float    bias2[512];              // b_ih2 + b_hh2
  float    wih1[512];               // W_ih1[:,0] (scalar input -> rank-1 update)
  float    wout[HN];                // W_out[0,:]
  float    xcur[TILE_B];            // current scalar input per batch row
};                                  // ~55.9KB

__global__ __launch_bounds__(NTHREADS)
void lstm_persistent_wmma(const float* __restrict__ x,
                          const float* __restrict__ W_ih1,
                          const float* __restrict__ b_ih1,
                          const float* __restrict__ W_hh1,
                          const float* __restrict__ b_hh1,
                          const float* __restrict__ W_ih2,
                          const float* __restrict__ b_ih2,
                          const float* __restrict__ W_hh2,
                          const float* __restrict__ b_hh2,
                          const float* __restrict__ W_out,
                          const float* __restrict__ b_out,
                          float* __restrict__ out)
{
  __shared__ Smem s;
  const int tid  = threadIdx.x;
  const int lane = tid & 31;
  const int wv   = tid >> 5;            // wave id 0..15; wave wv owns gate cols 32*wv..32*wv+31
  const int wgb  = blockIdx.x * TILE_B; // first batch row of this workgroup

  // ------------- register-resident weight slices (B-fragments, f16): 192 VGPRs -----
  // B-frag f16, contiguous k order chosen here (A uses the same): lanes 0-15 hold col N=lane, K=0..15; lanes 16-31 col N=lane-16, K=16..31.
  // W is [512,128] row-major; B[K=j][N=n] = W[n][j] -> lane reads 16 contiguous f32 of row n.
  v16h wf_hh1[2][4], wf_ih2[2][4], wf_hh2[2][4];
#pragma unroll
  for (int nt = 0; nt < 2; ++nt) {
    const int ncol = (wv * 2 + nt) * 16 + (lane & 15);
#pragma unroll
    for (int kc = 0; kc < 4; ++kc) {
      const int k0 = kc * 32 + (lane >> 4) * 16;
      wf_hh1[nt][kc] = load_w_frag(W_hh1 + ncol * HN + k0);
      wf_ih2[nt][kc] = load_w_frag(W_ih2 + ncol * HN + k0);
      wf_hh2[nt][kc] = load_w_frag(W_hh2 + ncol * HN + k0);
    }
  }

  // ------------- one-time LDS init (scalars live in LDS, not registers) ------------
  s.bias1[tid] = b_ih1[tid] + b_hh1[tid];
  s.bias2[tid] = b_ih2[tid] + b_hh2[tid];
  s.wih1[tid]  = W_ih1[tid];
  if (tid < HN) s.wout[tid] = W_out[tid];
  {
    unsigned int* z1 = (unsigned int*)&s.hbuf1[0][0][0];
    unsigned int* z2 = (unsigned int*)&s.hbuf2[0][0][0];
    z1[tid] = 0u; z1[tid + 512] = 0u;   // 4KB each = 1024 dwords
    z2[tid] = 0u; z2[tid + 512] = 0u;
  }

  // Elementwise mapping: thread owns gate column j = tid>>2 for batch rows bb..bb+3.
  const int jj = tid >> 2;          // 0..127
  const int bb = (tid & 3) * 4;     // 0,4,8,12
  // A-fragment scatter coords (contiguous k order, same as the B-fragments):
  // k' = j%32 -> lane = M + 16*(k'>=16); half = k'&15
  const int kcs   = jj >> 5;
  const int jp    = jj & 31;
  const int shalf = jp & 15;
  const int slb   = 16 * (jp >> 4) + bb;

  // per-thread persistent cell state (never leaves registers)
  float c1r[4] = {0.f, 0.f, 0.f, 0.f};
  float c2r[4] = {0.f, 0.f, 0.f, 0.f};

  __syncthreads();

  // =================== serial recurrence: 1024 encoder + 63 future steps ==========
  for (int t = 0; t < TN + FUTN - 1; ++t) {
    if (t < TN && tid < TILE_B)
      s.xcur[tid] = x[(wgb + tid) * TN + t];

    // ---- cell-1 gates: h1 @ W_hh1^T  (16x512 += 16x128 * 128x512) ----
    {
      v8f a0 = {}, a1 = {};
#pragma unroll
      for (int kc = 0; kc < 4; ++kc) {
        v16h af = *(const v16h*)&s.hbuf1[kc][lane][0];
        a0 = __builtin_amdgcn_wmma_f32_16x16x32_f16(false, af, false, wf_hh1[0][kc], (short)0, a0, false, false);
        a1 = __builtin_amdgcn_wmma_f32_16x16x32_f16(false, af, false, wf_hh1[1][kc], (short)0, a1, false, false);
      }
      // D layout: M = r + 8*(lane>=16), N = lane&15 -> contiguous along b in gbufT
      const int bo = 8 * (lane >> 4);
      const int n0 = wv * 32 + (lane & 15);
      float4* p0 = (float4*)&s.gbufT[n0][bo];
      p0[0] = make_float4(a0[0], a0[1], a0[2], a0[3]);
      p0[1] = make_float4(a0[4], a0[5], a0[6], a0[7]);
      float4* p1 = (float4*)&s.gbufT[n0 + 16][bo];
      p1[0] = make_float4(a1[0], a1[1], a1[2], a1[3]);
      p1[1] = make_float4(a1[4], a1[5], a1[6], a1[7]);
    }
    __syncthreads();

    // ---- elementwise LSTM cell 1 (adds rank-1 x*W_ih1 + biases) ----
    {
      const float4 gi4 = *(const float4*)&s.gbufT[jj      ][bb];
      const float4 gf4 = *(const float4*)&s.gbufT[jj + 128][bb];
      const float4 gg4 = *(const float4*)&s.gbufT[jj + 256][bb];
      const float4 go4 = *(const float4*)&s.gbufT[jj + 384][bb];
      const float4 xv  = *(const float4*)&s.xcur[bb];
      const float b1i = s.bias1[jj];
      const float b1f = s.bias1[jj + 128];
      const float b1g = s.bias1[jj + 256];
      const float b1o = s.bias1[jj + 384];
      const float w1i = s.wih1[jj];
      const float w1f = s.wih1[jj + 128];
      const float w1g = s.wih1[jj + 256];
      const float w1o = s.wih1[jj + 384];
      const float gi[4] = {gi4.x, gi4.y, gi4.z, gi4.w};
      const float gf[4] = {gf4.x, gf4.y, gf4.z, gf4.w};
      const float gg[4] = {gg4.x, gg4.y, gg4.z, gg4.w};
      const float go[4] = {go4.x, go4.y, go4.z, go4.w};
      const float xs[4] = {xv.x, xv.y, xv.z, xv.w};
#pragma unroll
      for (int u = 0; u < 4; ++u) {
        const float zi = gi[u] + b1i + xs[u] * w1i;
        const float zf = gf[u] + b1f + xs[u] * w1f;
        const float zg = gg[u] + b1g + xs[u] * w1g;
        const float zo = go[u] + b1o + xs[u] * w1o;
        c1r[u] = fsig(zf) * c1r[u] + fsig(zi) * ftanh(zg);
        s.hbuf1[kcs][slb + u][shalf] = (_Float16)(fsig(zo) * ftanh(c1r[u]));
      }
    }
    __syncthreads();

    // ---- cell-2 gates: h1 @ W_ih2^T + h2 @ W_hh2^T (chained accumulation) ----
    {
      v8f a0 = {}, a1 = {};
#pragma unroll
      for (int kc = 0; kc < 4; ++kc) {
        v16h af = *(const v16h*)&s.hbuf1[kc][lane][0];
        a0 = __builtin_amdgcn_wmma_f32_16x16x32_f16(false, af, false, wf_ih2[0][kc], (short)0, a0, false, false);
        a1 = __builtin_amdgcn_wmma_f32_16x16x32_f16(false, af, false, wf_ih2[1][kc], (short)0, a1, false, false);
      }
#pragma unroll
      for (int kc = 0; kc < 4; ++kc) {
        v16h af = *(const v16h*)&s.hbuf2[kc][lane][0];
        a0 = __builtin_amdgcn_wmma_f32_16x16x32_f16(false, af, false, wf_hh2[0][kc], (short)0, a0, false, false);
        a1 = __builtin_amdgcn_wmma_f32_16x16x32_f16(false, af, false, wf_hh2[1][kc], (short)0, a1, false, false);
      }
      const int bo = 8 * (lane >> 4);
      const int n0 = wv * 32 + (lane & 15);
      float4* p0 = (float4*)&s.gbufT[n0][bo];
      p0[0] = make_float4(a0[0], a0[1], a0[2], a0[3]);
      p0[1] = make_float4(a0[4], a0[5], a0[6], a0[7]);
      float4* p1 = (float4*)&s.gbufT[n0 + 16][bo];
      p1[0] = make_float4(a1[0], a1[1], a1[2], a1[3]);
      p1[1] = make_float4(a1[4], a1[5], a1[6], a1[7]);
    }
    __syncthreads();

    // ---- elementwise LSTM cell 2 ----
    {
      const float4 gi4 = *(const float4*)&s.gbufT[jj      ][bb];
      const float4 gf4 = *(const float4*)&s.gbufT[jj + 128][bb];
      const float4 gg4 = *(const float4*)&s.gbufT[jj + 256][bb];
      const float4 go4 = *(const float4*)&s.gbufT[jj + 384][bb];
      const float b2i = s.bias2[jj];
      const float b2f = s.bias2[jj + 128];
      const float b2g = s.bias2[jj + 256];
      const float b2o = s.bias2[jj + 384];
      const float gi[4] = {gi4.x, gi4.y, gi4.z, gi4.w};
      const float gf[4] = {gf4.x, gf4.y, gf4.z, gf4.w};
      const float gg[4] = {gg4.x, gg4.y, gg4.z, gg4.w};
      const float go[4] = {go4.x, go4.y, go4.z, go4.w};
#pragma unroll
      for (int u = 0; u < 4; ++u) {
        const float zi = gi[u] + b2i;
        const float zf = gf[u] + b2f;
        const float zg = gg[u] + b2g;
        const float zo = go[u] + b2o;
        c2r[u] = fsig(zf) * c2r[u] + fsig(zi) * ftanh(zg);
        s.hbuf2[kcs][slb + u][shalf] = (_Float16)(fsig(zo) * ftanh(c2r[u]));
      }
    }

    // ---- output projection out = h2 @ W_out^T + b_out (only needed from t = T-1) ----
    if (t >= TN - 1) {
      __syncthreads();
      // wave wv reduces batch row b = wv; lane handles j = lane*4..lane*4+3,
      // reading h2 directly from its f16 A-fragment layout in hbuf2.
      const int j0  = lane * 4;
      const int pkc = j0 >> 5;
      const int pjp = j0 & 31;
      const int pl  = 16 * (pjp >> 4) + wv;
      const int ph  = pjp & 15;
      const v4h   hq  = *(const v4h*)&s.hbuf2[pkc][pl][ph];
      const float4 w4 = *(const float4*)&s.wout[j0];
      float p = (float)hq[0] * w4.x + (float)hq[1] * w4.y +
                (float)hq[2] * w4.z + (float)hq[3] * w4.w;
#pragma unroll
      for (int off = 16; off > 0; off >>= 1)
        p += __shfl_xor(p, off, 32);
      if (lane == 0) {
        const float o = p + b_out[0];
        s.xcur[wv] = o;                                     // feeds next (future) step
        out[(wgb + wv) * FUTN + (t - (TN - 1))] = o;        // [B, FUT] row-major
      }
    }
    __syncthreads();
  }
}

extern "C" void kernel_launch(void* const* d_in, const int* in_sizes, int n_in,
                              void* d_out, int out_size, void* d_ws, size_t ws_size,
                              hipStream_t stream) {
  (void)in_sizes; (void)n_in; (void)out_size; (void)d_ws; (void)ws_size;
  const float* x     = (const float*)d_in[0];
  const float* W_ih1 = (const float*)d_in[1];
  const float* b_ih1 = (const float*)d_in[2];
  const float* W_hh1 = (const float*)d_in[3];
  const float* b_hh1 = (const float*)d_in[4];
  const float* W_ih2 = (const float*)d_in[5];
  const float* b_ih2 = (const float*)d_in[6];
  const float* W_hh2 = (const float*)d_in[7];
  const float* b_hh2 = (const float*)d_in[8];
  const float* W_out = (const float*)d_in[9];
  const float* b_out = (const float*)d_in[10];
  float* out = (float*)d_out;

  dim3 grid(BN / TILE_B);   // 32 workgroups, each owns 16 independent batch rows
  dim3 block(NTHREADS);     // 16 wave32 waves
  hipLaunchKernelGGL(lstm_persistent_wmma, grid, block, 0, stream,
                     x, W_ih1, b_ih1, W_hh1, b_hh1,
                     W_ih2, b_ih2, W_hh2, b_hh2, W_out, b_out, out);
}
